// SchmidhuberLinear_65644280152104
// MI455X (gfx1250) — matrix operand order repaired and checked by reading, not compiled, not run on hardware
//
#include <hip/hip_runtime.h>

// Problem constants (match reference)
#define B_DIM  64
#define D_DIM  2048
#define E_DIM  64
#define H1_DIM 512
#define H2_DIM 256
#define EPS_F  1e-5f

typedef __attribute__((ext_vector_type(16))) _Float16 v16h;
typedef __attribute__((ext_vector_type(8)))  float    v8f;
typedef __attribute__((ext_vector_type(4)))  _Float16 h4;

union Frag16 { v16h v; uint4 q[2]; };

// ---------------------------------------------------------------------------
// K0: fold BN constants.  s1=g1*rsqrt(v1+eps), t1=beta1-m1*s1
//                         s2=g2*rsqrt(v2+eps), t2=beta2+(b2-m2)*s2
// ---------------------------------------------------------------------------
__global__ __launch_bounds__(512) void k_consts(
    const float* __restrict__ g1, const float* __restrict__ v1,
    const float* __restrict__ beta1, const float* __restrict__ m1,
    const float* __restrict__ g2, const float* __restrict__ v2,
    const float* __restrict__ beta2, const float* __restrict__ m2,
    const float* __restrict__ b2,
    float* __restrict__ s1, float* __restrict__ t1,
    float* __restrict__ s2, float* __restrict__ t2)
{
    int i = threadIdx.x;
    if (i < H1_DIM) {
        float s = g1[i] * rsqrtf(v1[i] + EPS_F);
        s1[i] = s;
        t1[i] = beta1[i] - m1[i] * s;
    }
    if (i < H2_DIM) {
        float s = g2[i] * rsqrtf(v2[i] + EPS_F);
        s2[i] = s;
        t2[i] = beta2[i] + (b2[i] - m2[i]) * s;
    }
}

// ---------------------------------------------------------------------------
// K1: P[b,h] = (rep[b,:] @ W1d[:,h] + b1[h]) * s1[h] + t1[h]   (64 x 512)
// ---------------------------------------------------------------------------
__global__ __launch_bounds__(256) void k_P(
    const float* __restrict__ rep, const float* __restrict__ W1,
    const float* __restrict__ b1, const float* __restrict__ s1,
    const float* __restrict__ t1, float* __restrict__ P)
{
    int idx = blockIdx.x * 256 + threadIdx.x;   // 0..32767
    int b = idx >> 9;
    int h = idx & (H1_DIM - 1);
    const float* r = rep + (size_t)b * D_DIM;
    float acc = 0.f;
#pragma unroll 8
    for (int d = 0; d < D_DIM; ++d)
        acc = fmaf(r[d], W1[(size_t)d * H1_DIM + h], acc);
    P[idx] = (acc + b1[h]) * s1[h] + t1[h];
}

// ---------------------------------------------------------------------------
// K2: Q[d,h] = (emb[d,:] @ W1e[:,h]) * s1[h]   (2048 x 512)
// ---------------------------------------------------------------------------
__global__ __launch_bounds__(256) void k_Q(
    const float* __restrict__ emb, const float* __restrict__ W1,
    const float* __restrict__ s1, float* __restrict__ Q)
{
    int idx = blockIdx.x * 256 + threadIdx.x;   // 0..1048575
    int d = idx >> 9;
    int h = idx & (H1_DIM - 1);
    const float* er = emb + (size_t)d * E_DIM;
    float acc = 0.f;
#pragma unroll
    for (int e = 0; e < E_DIM; ++e)
        acc = fmaf(er[e], W1[(size_t)(D_DIM + e) * H1_DIM + h], acc);
    Q[idx] = acc * s1[h];
}

// ---------------------------------------------------------------------------
// K3: W2T[o,h] = (f16) W2[h,o]   (256 x 512, row-major over K=H1)
// ---------------------------------------------------------------------------
__global__ __launch_bounds__(256) void k_w2t(
    const float* __restrict__ W2, _Float16* __restrict__ W2T)
{
    int idx = blockIdx.x * 256 + threadIdx.x;   // 0..131071
    int o = idx & (H2_DIM - 1);                 // coalesced read side
    int h = idx >> 8;
    W2T[(size_t)o * H1_DIM + h] = (_Float16)W2[(size_t)h * H2_DIM + o];
}

// ---------------------------------------------------------------------------
// Main fused kernel.
// Per block: 64 rows m = b*D + d (same b), full N = 256.
// 8 waves; wave w owns M-tiles {2*(w>>2), 2*(w>>2)+1} x N-quadrant (w&3)*64:
//   => 8 x v_wmma_f32_16x16x32_f16 per K-step per wave.
// Double-buffered f16 A tile in LDS: ONE barrier per K-step.
//   h1[m,h] = relu(P[b,h] + Q[d,h] - rep[b,d]*W1d[d,h]*s1[h])
//   out[m]  = sum_o relu((h1 @ W2)[o]*s2[o]+t2[o])*W3[o] + b3
// ---------------------------------------------------------------------------
__global__ __launch_bounds__(256) void k_main(
    const float* __restrict__ rep, const float* __restrict__ W1,
    const float* __restrict__ Qm,  const float* __restrict__ Pm,
    const float* __restrict__ s1g, const float* __restrict__ s2g,
    const float* __restrict__ t2g, const float* __restrict__ W3g,
    const float* __restrict__ b3g, const _Float16* __restrict__ W2T,
    float* __restrict__ out)
{
    __shared__ __align__(16) float    Prow[H1_DIM];
    __shared__ __align__(16) float    S1[H1_DIM];
    __shared__ __align__(16) float    S2[H2_DIM];
    __shared__ __align__(16) float    T2[H2_DIM];
    __shared__ __align__(16) float    W3s[H2_DIM];
    __shared__ __align__(16) float    RepS[64];
    __shared__ __align__(16) _Float16 Atile[2][64 * 40]; // 64 rows x 32 K, stride 40
    __shared__ __align__(16) float    OutAcc[64];

    const int t  = threadIdx.x;
    const int m0 = blockIdx.x * 64;          // flattened (b,d) row base
    const int b  = m0 >> 11;                 // / D_DIM
    const int d0 = m0 & (D_DIM - 1);

    // ---- preload block-invariant data into LDS ----
    Prow[t]       = Pm[(size_t)b * H1_DIM + t];
    Prow[t + 256] = Pm[(size_t)b * H1_DIM + 256 + t];
    S1[t]         = s1g[t];
    S1[t + 256]   = s1g[t + 256];
    if (t < H2_DIM) { S2[t] = s2g[t]; T2[t] = t2g[t]; W3s[t] = W3g[t]; }
    if (t < 64)     { RepS[t] = rep[(size_t)b * D_DIM + d0 + t]; OutAcc[t] = 0.f; }
    __syncthreads();

    const int wave = t >> 5;
    const int lane = t & 31;
    const int mp   = wave >> 2;              // 32-row M half (0/1)
    const int nq   = wave & 3;               // 64-column N quadrant
    const int ln   = lane & 15;
    const int lh   = lane >> 4;

    // A-tile build mapping: 2 rows x 4 consecutive h per thread
    const int arow0 = t >> 3;                // 0..31
    const int arow1 = arow0 + 32;            // 32..63
    const int acol  = (t & 7) << 2;          // 0,4,...,28
    const float repv0 = RepS[arow0];
    const float repv1 = RepS[arow1];
    const float* qrow0 = Qm + (size_t)(d0 + arow0) * H1_DIM;
    const float* wrow0 = W1 + (size_t)(d0 + arow0) * H1_DIM;
    const float* qrow1 = Qm + (size_t)(d0 + arow1) * H1_DIM;
    const float* wrow1 = W1 + (size_t)(d0 + arow1) * H1_DIM;
    const int ast0 = arow0 * 40 + acol;
    const int ast1 = arow1 * 40 + acol;

    // h1 tile element builder (4 consecutive h)
    auto build = [&](const float* qr, const float* wr, float rv, int kb) -> h4 {
        float4 q4 = *(const float4*)(qr + kb + acol);
        float4 w4 = *(const float4*)(wr + kb + acol);
        float4 s4 = *(const float4*)(S1 + kb + acol);
        float4 p4 = *(const float4*)(Prow + kb + acol);
        float e0 = fmaxf(p4.x + q4.x - rv * w4.x * s4.x, 0.f);
        float e1 = fmaxf(p4.y + q4.y - rv * w4.y * s4.y, 0.f);
        float e2 = fmaxf(p4.z + q4.z - rv * w4.z * s4.z, 0.f);
        float e3 = fmaxf(p4.w + q4.w - rv * w4.w * s4.w, 0.f);
        h4 hv;
        hv.x = (_Float16)e0; hv.y = (_Float16)e1;
        hv.z = (_Float16)e2; hv.w = (_Float16)e3;
        return hv;
    };

    // A fragment offsets (ISA 16-bit A 16x32 layout):
    //   lanes 0-15 : row=lane,    K 0-7 (@+0)   and 16-23 (@+32B)
    //   lanes16-31 : row=lane-16, K 8-15 (@+16B) and 24-31 (@+48B)
    const int aoffA = (mp * 32 + ln) * 40 + lh * 8;       // M-tile 2*mp
    const int aoffB = (mp * 32 + 16 + ln) * 40 + lh * 8;  // M-tile 2*mp+1

    // B fragment base (ISA 16-bit B 32x16 layout): lane = column n,
    //   lanes 0-15 hold K 0-7,16-23, lanes 16-31 hold K 8-15,24-31
    const _Float16* bbase = W2T + (size_t)(nq * 64 + ln) * H1_DIM + lh * 8;

    v8f accA0 = {}, accA1 = {}, accA2 = {}, accA3 = {};
    v8f accB0 = {}, accB1 = {}, accB2 = {}, accB3 = {};

    // ---- prologue: build K-step 0 into buffer 0 ----
    {
        h4 a0 = build(qrow0, wrow0, repv0, 0);
        h4 a1 = build(qrow1, wrow1, repv1, 0);
        *(h4*)(&Atile[0][ast0]) = a0;
        *(h4*)(&Atile[0][ast1]) = a1;
    }
    __syncthreads();

    for (int ks = 0; ks < H1_DIM / 32; ++ks) {
        const int kb  = ks << 5;
        const int cur = ks & 1;
        const bool have_next = (ks + 1) < (H1_DIM / 32);

        // --- start next tile's h1 computation (global loads issue early) ---
        h4 n0, n1;
        if (have_next) {
            __builtin_prefetch(qrow0 + kb + 64 + acol, 0, 3);
            __builtin_prefetch(wrow0 + kb + 64 + acol, 0, 3);
            n0 = build(qrow0, wrow0, repv0, kb + 32);
            n1 = build(qrow1, wrow1, repv1, kb + 32);
        }

        // --- fragment loads for current K-step ---
        const _Float16* ab = &Atile[cur][0];
        Frag16 afA, afB;
        afA.q[0] = *(const uint4*)(ab + aoffA);
        afA.q[1] = *(const uint4*)(ab + aoffA + 16);
        afB.q[0] = *(const uint4*)(ab + aoffB);
        afB.q[1] = *(const uint4*)(ab + aoffB + 16);

        const _Float16* bp = bbase + kb;
        Frag16 bf0, bf1, bf2, bf3;
        bf0.q[0] = *(const uint4*)(bp);
        bf0.q[1] = *(const uint4*)(bp + 16);
        bf1.q[0] = *(const uint4*)(bp + 16 * H1_DIM);
        bf1.q[1] = *(const uint4*)(bp + 16 * H1_DIM + 16);
        bf2.q[0] = *(const uint4*)(bp + 32 * H1_DIM);
        bf2.q[1] = *(const uint4*)(bp + 32 * H1_DIM + 16);
        bf3.q[0] = *(const uint4*)(bp + 48 * H1_DIM);
        bf3.q[1] = *(const uint4*)(bp + 48 * H1_DIM + 16);

        // --- 8 WMMAs: 2 M-tiles x 4 N-tiles, B fragments reused twice ---
        accA0 = __builtin_amdgcn_wmma_f32_16x16x32_f16(false, afA.v, false, bf0.v, (short)0, accA0, false, false);
        accB0 = __builtin_amdgcn_wmma_f32_16x16x32_f16(false, afB.v, false, bf0.v, (short)0, accB0, false, false);
        accA1 = __builtin_amdgcn_wmma_f32_16x16x32_f16(false, afA.v, false, bf1.v, (short)0, accA1, false, false);
        accB1 = __builtin_amdgcn_wmma_f32_16x16x32_f16(false, afB.v, false, bf1.v, (short)0, accB1, false, false);
        accA2 = __builtin_amdgcn_wmma_f32_16x16x32_f16(false, afA.v, false, bf2.v, (short)0, accA2, false, false);
        accB2 = __builtin_amdgcn_wmma_f32_16x16x32_f16(false, afB.v, false, bf2.v, (short)0, accB2, false, false);
        accA3 = __builtin_amdgcn_wmma_f32_16x16x32_f16(false, afA.v, false, bf3.v, (short)0, accA3, false, false);
        accB3 = __builtin_amdgcn_wmma_f32_16x16x32_f16(false, afB.v, false, bf3.v, (short)0, accB3, false, false);

        // --- store next tile into the other buffer (no conflict w/ reads) ---
        if (have_next) {
            _Float16* an = &Atile[cur ^ 1][0];
            *(h4*)(&an[ast0]) = n0;
            *(h4*)(&an[ast1]) = n1;
        }
        __syncthreads();   // one barrier per K-step (double buffered)
    }

    // ---- epilogue: BN2 + ReLU + dot(W3), reduce across N ----
    // C layout: VGPR j, lane l -> M = j + 8*(l>>4), N = (l&15)
#pragma unroll
    for (int j = 0; j < 8; ++j) {
        const int n = nq * 64 + ln;
        float v, sA, sB;
        v = fmaf(accA0[j], S2[n],      T2[n]);      v = fmaxf(v, 0.f); sA = v * W3s[n];
        v = fmaf(accA1[j], S2[n + 16], T2[n + 16]); v = fmaxf(v, 0.f); sA = fmaf(v, W3s[n + 16], sA);
        v = fmaf(accA2[j], S2[n + 32], T2[n + 32]); v = fmaxf(v, 0.f); sA = fmaf(v, W3s[n + 32], sA);
        v = fmaf(accA3[j], S2[n + 48], T2[n + 48]); v = fmaxf(v, 0.f); sA = fmaf(v, W3s[n + 48], sA);
        v = fmaf(accB0[j], S2[n],      T2[n]);      v = fmaxf(v, 0.f); sB = v * W3s[n];
        v = fmaf(accB1[j], S2[n + 16], T2[n + 16]); v = fmaxf(v, 0.f); sB = fmaf(v, W3s[n + 16], sB);
        v = fmaf(accB2[j], S2[n + 32], T2[n + 32]); v = fmaxf(v, 0.f); sB = fmaf(v, W3s[n + 32], sB);
        v = fmaf(accB3[j], S2[n + 48], T2[n + 48]); v = fmaxf(v, 0.f); sB = fmaf(v, W3s[n + 48], sB);
        // reduce across the 16 lanes that share each output row
        sA += __shfl_xor(sA, 1, 32);
        sA += __shfl_xor(sA, 2, 32);
        sA += __shfl_xor(sA, 4, 32);
        sA += __shfl_xor(sA, 8, 32);
        sB += __shfl_xor(sB, 1, 32);
        sB += __shfl_xor(sB, 2, 32);
        sB += __shfl_xor(sB, 4, 32);
        sB += __shfl_xor(sB, 8, 32);
        if (ln == 0) {
            atomicAdd(&OutAcc[mp * 32 + lh * 8 + j], sA);
            atomicAdd(&OutAcc[mp * 32 + 16 + lh * 8 + j], sB);
        }
    }

    __syncthreads();
    if (t < 64) out[m0 + t] = OutAcc[t] + b3g[0];
}

// ---------------------------------------------------------------------------
// Launch
// ---------------------------------------------------------------------------
extern "C" void kernel_launch(void* const* d_in, const int* in_sizes, int n_in,
                              void* d_out, int out_size, void* d_ws, size_t ws_size,
                              hipStream_t stream)
{
    const float* rep   = (const float*)d_in[0];
    const float* emb   = (const float*)d_in[1];
    const float* W1    = (const float*)d_in[2];
    const float* b1    = (const float*)d_in[3];
    const float* g1    = (const float*)d_in[4];
    const float* beta1 = (const float*)d_in[5];
    const float* m1    = (const float*)d_in[6];
    const float* v1    = (const float*)d_in[7];
    const float* W2    = (const float*)d_in[8];
    const float* b2    = (const float*)d_in[9];
    const float* g2    = (const float*)d_in[10];
    const float* beta2 = (const float*)d_in[11];
    const float* m2    = (const float*)d_in[12];
    const float* v2    = (const float*)d_in[13];
    const float* W3    = (const float*)d_in[14];
    const float* b3    = (const float*)d_in[15];
    float* out = (float*)d_out;

    // workspace (floats): s1[512] t1[512] s2[256] t2[256] P[64*512] Q[2048*512] | W2T f16[256*512]
    float* ws = (float*)d_ws;
    float* s1 = ws;
    float* t1 = ws + 512;
    float* s2 = ws + 1024;
    float* t2 = ws + 1280;
    float* P  = ws + 1536;
    float* Q  = ws + 1536 + B_DIM * H1_DIM;
    _Float16* W2T = (_Float16*)(ws + 1536 + B_DIM * H1_DIM + D_DIM * H1_DIM);
    // total: ~4.6 MB

    k_consts<<<1, 512, 0, stream>>>(g1, v1, beta1, m1, g2, v2, beta2, m2, b2,
                                    s1, t1, s2, t2);
    k_P   <<<(B_DIM * H1_DIM) / 256, 256, 0, stream>>>(rep, W1, b1, s1, t1, P);
    k_Q   <<<(D_DIM * H1_DIM) / 256, 256, 0, stream>>>(emb, W1, s1, Q);
    k_w2t <<<(H1_DIM * H2_DIM) / 256, 256, 0, stream>>>(W2, W2T);
    k_main<<<(B_DIM * D_DIM) / 64,   256, 0, stream>>>(rep, W1, Q, P, s1, s2, t2,
                                                       W3, b3, W2T, out);
}
